// LSTMKeyAttn_163208757253
// MI455X (gfx1250) — compile-verified, matrix operand order checked by reading, not run on hardware
//
#include <hip/hip_runtime.h>
#include <hip/hip_bf16.h>

// ---------------------------------------------------------------------------
// Problem constants (from reference)
// ---------------------------------------------------------------------------
#define BB   256      // batch
#define SS   160      // sequence length
#define EE   300      // embedding dim
#define HH   256      // hidden
#define TT   200      // key dim
#define G4   1024     // 4*H
#define KCAT 512      // padded E+T (300+200 -> 512)
#define MROW (BB*SS)  // 40960

typedef __attribute__((ext_vector_type(16))) __bf16 v16bf;
typedef __attribute__((ext_vector_type(8)))  float  v8f;

// ---------------------------------------------------------------------------
// WMMA fragment loaders (CDNA5 wave32 layouts, 16x16x32 bf16)
// A: 16(M) x 32(K) row-major, per CDNA5 ISA 16-bit A layout.
//   lane L: m = L%16, half = L/16
//   VGPR v (pair of bf16): k = (v<4 ? 0 : 16) + half*8 + 2*(v&3)
// B: 32(K) x 16(N), sourced from row-major W (N x K): B[k][n] = W[n*ldw + k]
//   same k mapping with n in the lane role.
// C/D (f32): lane L: n = L%16 ; VGPR v: m = v + 8*(L/16)
// ---------------------------------------------------------------------------
static __device__ inline v16bf load_a_frag(const __bf16* __restrict__ A, int lda) {
  const int lane = threadIdx.x & 31;
  const int m = lane & 15, half = lane >> 4;
  v16bf a;
#pragma unroll
  for (int v = 0; v < 8; ++v) {
    const int k = ((v & 4) << 2) + half * 8 + ((v & 3) << 1);
    a[2 * v]     = A[(size_t)m * lda + k];
    a[2 * v + 1] = A[(size_t)m * lda + k + 1];
  }
  return a;
}

static __device__ inline v16bf load_b_fragT(const __bf16* __restrict__ W, int ldw) {
  const int lane = threadIdx.x & 31;
  const int n = lane & 15, half = lane >> 4;
  v16bf b;
#pragma unroll
  for (int v = 0; v < 8; ++v) {
    const int k = ((v & 4) << 2) + half * 8 + ((v & 3) << 1);
    b[2 * v]     = W[(size_t)n * ldw + k];
    b[2 * v + 1] = W[(size_t)n * ldw + k + 1];
  }
  return b;
}

static __device__ inline v8f wmma_bf16(v16bf a, v16bf b, v8f c) {
  return __builtin_amdgcn_wmma_f32_16x16x32_bf16(
      /*neg_a=*/false, a, /*neg_b=*/false, b,
      /*c_mod=*/(short)0, c, /*reuse_a=*/false, /*reuse_b=*/false);
}

static __device__ inline float sigmoidf_fast(float x) {
  return 1.0f / (1.0f + __expf(-x));
}

// ---------------------------------------------------------------------------
// Utility kernels
// ---------------------------------------------------------------------------
__global__ void zero32_k(unsigned int* __restrict__ p, int n) {
  int i = blockIdx.x * blockDim.x + threadIdx.x;
  if (i < n) p[i] = 0u;
}

__global__ void f2bf_k(const float* __restrict__ in, __bf16* __restrict__ out, int n) {
  int i = blockIdx.x * blockDim.x + threadIdx.x;
  if (i < n) out[i] = (__bf16)in[i];
}

// Build Wcat (G4 x KCAT) bf16 = [W_ih | W_kh | 0-pad]
__global__ void prep_wcat_k(const float* __restrict__ W_ih,
                            const float* __restrict__ W_kh,
                            __bf16* __restrict__ Wcat) {
  const int row = blockIdx.x;           // 0..G4-1
  __bf16* out = Wcat + (size_t)row * KCAT;
  const float* wi = W_ih + (size_t)row * EE;
  const float* wk = W_kh + (size_t)row * TT;
  for (int c = threadIdx.x; c < KCAT; c += blockDim.x) {
    float v = 0.0f;
    if (c < EE)            v = wi[c];
    else if (c < EE + TT)  v = wk[c - EE];
    out[c] = (__bf16)v;
  }
}

// Embedding lookup + key concat + zero pad -> cat (MROW x KCAT) bf16
__global__ void embed_cat_k(const int* __restrict__ ids,
                            const float* __restrict__ key,       // (B,S,T)
                            const float* __restrict__ emb,       // (V,E)
                            __bf16* __restrict__ cat) {
  const int row = blockIdx.x;           // b*SS + s
  const int tok = ids[row];
  const float* e  = emb + (size_t)tok * EE;
  const float* kk = key + (size_t)row * TT;
  __bf16* out = cat + (size_t)row * KCAT;
  for (int c = threadIdx.x; c < KCAT; c += blockDim.x) {
    float v = 0.0f;
    if (c < EE)            v = e[c];
    else if (c < EE + TT)  v = kk[c - EE];
    out[c] = (__bf16)v;
  }
}

// ---------------------------------------------------------------------------
// Big input-projection GEMM: Out(MROW x G4) = cat(MROW x KCAT) @ Wcat^T + b_g
// Register-blocked 16(M) x 64(N) strip per wave + software-pipelined K loop:
// the next K-chunk's A/B fragments are fetched before the current chunk's
// 4 WMMAs issue, so VMEM overlaps the XDL pipe (partial loadcnt waits).
// grid = (MROW/16, G4/64/8), block = 256 (8 waves)
// ---------------------------------------------------------------------------
__global__ void __launch_bounds__(256)
gemm_bias_k(const __bf16* __restrict__ A,     // (MROW, KCAT)
            const __bf16* __restrict__ W,     // (G4, KCAT) -> used transposed
            const float* __restrict__ bias,   // (G4)
            float* __restrict__ Out) {
  const int wave = threadIdx.x >> 5;
  const int tm = blockIdx.x;                       // M tile (16 rows)
  const int tn = blockIdx.y * 8 + wave;            // N strip (64 cols)
  const int lane = threadIdx.x & 31;
  const int n = lane & 15, half = lane >> 4;

  const __bf16* Arow = A + (size_t)tm * 16 * KCAT;
  const __bf16* Wrow = W + (size_t)tn * 64 * KCAT;

  v8f acc[4];
#pragma unroll
  for (int j = 0; j < 4; ++j) {
    const float bv = bias[tn * 64 + j * 16 + n];
#pragma unroll
    for (int v = 0; v < 8; ++v) acc[j][v] = bv;
  }

  // ---- software pipeline: prefetch chunk 0 ----
  v16bf a_cur = load_a_frag(Arow, KCAT);
  v16bf b_cur[4], b_nxt[4];
#pragma unroll
  for (int j = 0; j < 4; ++j)
    b_cur[j] = load_b_fragT(Wrow + (size_t)j * 16 * KCAT, KCAT);

#pragma unroll
  for (int k0 = 0; k0 < KCAT; k0 += 32) {
    v16bf a_nxt;
    if (k0 + 32 < KCAT) {               // prefetch next chunk (unrolled: static)
      a_nxt = load_a_frag(Arow + k0 + 32, KCAT);
#pragma unroll
      for (int j = 0; j < 4; ++j)
        b_nxt[j] = load_b_fragT(Wrow + (size_t)j * 16 * KCAT + k0 + 32, KCAT);
    }
#pragma unroll
    for (int j = 0; j < 4; ++j)
      acc[j] = wmma_bf16(a_cur, b_cur[j], acc[j]);
    a_cur = a_nxt;
#pragma unroll
    for (int j = 0; j < 4; ++j) b_cur[j] = b_nxt[j];
  }

#pragma unroll
  for (int j = 0; j < 4; ++j) {
#pragma unroll
    for (int v = 0; v < 8; ++v) {
      const int m = v + 8 * half;
      Out[(size_t)(tm * 16 + m) * G4 + tn * 64 + j * 16 + n] = acc[j][v];
    }
  }
}

// ---------------------------------------------------------------------------
// Fused LSTM step: G = Gx[:,s,:] + h @ W_hh^T ; gates ; update c,h
// One wave per (16 batch x 16 hidden) tile computing all 4 gate tiles;
// K loop software-pipelined like the big GEMM (this kernel sits on the
// sequential critical path, 320 launches).
// grid = (BB/16, HH/16), block = 32.
// ---------------------------------------------------------------------------
__global__ void __launch_bounds__(32)
lstm_step_k(const __bf16* __restrict__ hbf_in,   // (BB,HH) bf16
            __bf16* __restrict__ hbf_out,        // (BB,HH) bf16
            const __bf16* __restrict__ Whh,      // (G4,HH) bf16
            const float* __restrict__ Gx,        // (MROW,G4)
            float* __restrict__ cbuf,            // (BB,HH)
            float* __restrict__ hbuf,            // (BB,HH)
            float* __restrict__ sc,              // (BB,SS,HH)
            int s, int store_sc) {
  const int tb = blockIdx.x;   // batch tile
  const int tj = blockIdx.y;   // hidden tile
  const int lane = threadIdx.x & 31;
  const int n = lane & 15, half = lane >> 4;

  v8f acc[4];
  // init accumulators from the precomputed input projection
#pragma unroll
  for (int g = 0; g < 4; ++g) {
#pragma unroll
    for (int v = 0; v < 8; ++v) {
      const int m = v + 8 * half;
      const size_t row = (size_t)(tb * 16 + m) * SS + s;
      acc[g][v] = Gx[row * G4 + g * HH + tj * 16 + n];
    }
  }

  const __bf16* Arow = hbf_in + (size_t)tb * 16 * HH;
  const __bf16* Wbase = Whh + (size_t)tj * 16 * HH;

  // ---- software pipeline: prefetch chunk 0 ----
  v16bf a_cur = load_a_frag(Arow, HH);
  v16bf b_cur[4], b_nxt[4];
#pragma unroll
  for (int g = 0; g < 4; ++g)
    b_cur[g] = load_b_fragT(Wbase + (size_t)g * HH * HH, HH);

#pragma unroll
  for (int k0 = 0; k0 < HH; k0 += 32) {
    v16bf a_nxt;
    if (k0 + 32 < HH) {
      a_nxt = load_a_frag(Arow + k0 + 32, HH);
#pragma unroll
      for (int g = 0; g < 4; ++g)
        b_nxt[g] = load_b_fragT(Wbase + (size_t)g * HH * HH + k0 + 32, HH);
    }
#pragma unroll
    for (int g = 0; g < 4; ++g)
      acc[g] = wmma_bf16(a_cur, b_cur[g], acc[g]);
    a_cur = a_nxt;
#pragma unroll
    for (int g = 0; g < 4; ++g) b_cur[g] = b_nxt[g];
  }

  // gates: [i | f | g | o]
#pragma unroll
  for (int v = 0; v < 8; ++v) {
    const int m = v + 8 * half;
    const int idx = (tb * 16 + m) * HH + tj * 16 + n;
    const float iv = sigmoidf_fast(acc[0][v]);
    const float fv = sigmoidf_fast(acc[1][v]);
    const float gv = tanhf(acc[2][v]);
    const float ov = sigmoidf_fast(acc[3][v]);
    const float cn = fv * cbuf[idx] + iv * gv;
    const float hn = ov * tanhf(cn);
    cbuf[idx] = cn;
    hbuf[idx] = hn;
    hbf_out[idx] = (__bf16)hn;
    if (store_sc)
      sc[((size_t)(tb * 16 + m) * SS + s) * HH + tj * 16 + n] = hn;
  }
}

// ---------------------------------------------------------------------------
// Attention tail (algebraically reduced):
//   u[b]   = r[b] @ W_attn            (so energies = sc . u + b_attn . r)
//   e0[b]  = b_attn . r[b]
// ---------------------------------------------------------------------------
__global__ void attn_u_k(const float* __restrict__ r,
                         const float* __restrict__ W_attn,   // (H,H)
                         const float* __restrict__ b_attn,   // (H)
                         float* __restrict__ u, float* __restrict__ e0) {
  const int b = blockIdx.x, k = threadIdx.x;   // HH threads
  float acc = 0.0f;
  for (int h = 0; h < HH; ++h) acc += r[b * HH + h] * W_attn[h * HH + k];
  u[b * HH + k] = acc;
  __shared__ float red[HH];
  red[k] = b_attn[k] * r[b * HH + k];
  __syncthreads();
  for (int st = HH / 2; st > 0; st >>= 1) {
    if (k < st) red[k] += red[k + st];
    __syncthreads();
  }
  if (k == 0) e0[b] = red[0];
}

// energies[b,s] = (sc[b,s,:] . u[b,:] + e0[b]) * mask[b,s]
// grid = (BB, SS/8), block = 256 (8 waves, one s each)
__global__ void energies_k(const float* __restrict__ sc,
                           const float* __restrict__ u,
                           const float* __restrict__ e0,
                           const float* __restrict__ mask,
                           float* __restrict__ energ) {
  const int b = blockIdx.x;
  const int wave = threadIdx.x >> 5, lane = threadIdx.x & 31;
  const int s = blockIdx.y * 8 + wave;
  const float* scp = sc + ((size_t)b * SS + s) * HH;
  const float* up  = u + (size_t)b * HH;
  float acc = 0.0f;
  for (int h = lane; h < HH; h += 32) acc += scp[h] * up[h];
  for (int off = 16; off > 0; off >>= 1) acc += __shfl_xor(acc, off, 32);
  if (lane == 0) energ[b * SS + s] = (acc + e0[b]) * mask[b * SS + s];
}

__global__ void softmax_k(const float* __restrict__ energ,
                          float* __restrict__ alpha) {
  const int b = blockIdx.x, t = threadIdx.x;   // 256 threads, SS=160 active
  __shared__ float red[256];
  const float v = (t < SS) ? energ[b * SS + t] : -1e30f;
  red[t] = v; __syncthreads();
  for (int st = 128; st > 0; st >>= 1) {
    if (t < st) red[t] = fmaxf(red[t], red[t + st]);
    __syncthreads();
  }
  const float mx = red[0]; __syncthreads();
  const float ex = (t < SS) ? __expf(v - mx) : 0.0f;
  red[t] = ex; __syncthreads();
  for (int st = 128; st > 0; st >>= 1) {
    if (t < st) red[t] += red[t + st];
    __syncthreads();
  }
  const float inv = 1.0f / red[0];
  if (t < SS) alpha[b * SS + t] = ex * inv;
}

// c_attn[b,h] = sum_s alpha[b,s] * sc[b,s,h]
__global__ void cattn_k(const float* __restrict__ alpha,
                        const float* __restrict__ sc,
                        float* __restrict__ c_attn) {
  const int b = blockIdx.x, h = threadIdx.x;   // HH threads
  float acc = 0.0f;
  for (int s = 0; s < SS; ++s)
    acc += alpha[b * SS + s] * sc[((size_t)b * SS + s) * HH + h];
  c_attn[b * HH + h] = acc;
}

// o[b] = c_attn[b] . (M @ r[b]) + b0
__global__ void final_k(const float* __restrict__ c_attn,
                        const float* __restrict__ Mw,     // (H,H)
                        const float* __restrict__ r,
                        const float* __restrict__ b0,
                        float* __restrict__ o) {
  const int b = blockIdx.x, h = threadIdx.x;   // HH threads
  float t = 0.0f;
  for (int k = 0; k < HH; ++k) t += Mw[h * HH + k] * r[b * HH + k];
  __shared__ float red[HH];
  red[h] = c_attn[b * HH + h] * t;
  __syncthreads();
  for (int st = HH / 2; st > 0; st >>= 1) {
    if (h < st) red[h] += red[h + st];
    __syncthreads();
  }
  if (h == 0) o[b] = red[0] + b0[0];
}

// ---------------------------------------------------------------------------
// Host launcher
// ---------------------------------------------------------------------------
extern "C" void kernel_launch(void* const* d_in, const int* in_sizes, int n_in,
                              void* d_out, int out_size, void* d_ws, size_t ws_size,
                              hipStream_t stream) {
  (void)in_sizes; (void)n_in; (void)out_size; (void)ws_size;

  const int*   x1     = (const int*)  d_in[0];
  const int*   x2     = (const int*)  d_in[1];
  const float* x1mask = (const float*)d_in[2];
  const float* key_c  = (const float*)d_in[3];
  const float* key_r  = (const float*)d_in[4];
  const float* emb    = (const float*)d_in[5];
  const float* W_ih   = (const float*)d_in[6];
  const float* W_hh   = (const float*)d_in[7];
  const float* W_kh   = (const float*)d_in[8];
  const float* b_g    = (const float*)d_in[9];
  const float* W_attn = (const float*)d_in[10];
  const float* b_attn = (const float*)d_in[11];
  const float* Mw     = (const float*)d_in[12];
  const float* b0     = (const float*)d_in[13];
  float* out = (float*)d_out;

  // -------- workspace carve --------
  char* ws = (char*)d_ws;
  size_t off = 0;
  auto carve = [&](size_t bytes) -> void* {
    void* p = ws + off;
    off = (off + bytes + 255) & ~(size_t)255;
    return p;
  };
  __bf16* cat    = (__bf16*)carve((size_t)MROW * KCAT * 2);   // 42 MB (reused x2)
  __bf16* Wcat   = (__bf16*)carve((size_t)G4 * KCAT * 2);
  __bf16* WhhB   = (__bf16*)carve((size_t)G4 * HH * 2);
  float*  Gx     = (float*) carve((size_t)MROW * G4 * 4);     // 168 MB (reused x2)
  float*  sc     = (float*) carve((size_t)MROW * HH * 4);     // 42 MB (RNN1 states)
  float*  hbuf   = (float*) carve((size_t)BB * HH * 4);
  float*  cbuf   = (float*) carve((size_t)BB * HH * 4);
  __bf16* hbf0   = (__bf16*)carve((size_t)BB * HH * 2);
  __bf16* hbf1   = (__bf16*)carve((size_t)BB * HH * 2);
  float*  u      = (float*) carve((size_t)BB * HH * 4);
  float*  e0     = (float*) carve((size_t)BB * 4);
  float*  energ  = (float*) carve((size_t)BB * SS * 4);
  float*  alpha  = (float*) carve((size_t)BB * SS * 4);
  float*  cattn  = (float*) carve((size_t)BB * HH * 4);

  // -------- weight prep (bf16) --------
  prep_wcat_k<<<G4, 128, 0, stream>>>(W_ih, W_kh, Wcat);
  f2bf_k<<<(G4 * HH + 255) / 256, 256, 0, stream>>>(W_hh, WhhB, G4 * HH);

  // -------- one RNN pass --------
  auto run_rnn = [&](const int* ids, const float* key, int store_sc) {
    embed_cat_k<<<MROW, 128, 0, stream>>>(ids, key, emb, cat);
    gemm_bias_k<<<dim3(MROW / 16, G4 / 64 / 8), 256, 0, stream>>>(cat, Wcat, b_g, Gx);

    const int nstate = BB * HH;
    zero32_k<<<(nstate + 255) / 256, 256, 0, stream>>>((unsigned int*)hbuf, nstate);
    zero32_k<<<(nstate + 255) / 256, 256, 0, stream>>>((unsigned int*)cbuf, nstate);
    zero32_k<<<(nstate / 2 + 255) / 256, 256, 0, stream>>>((unsigned int*)hbf0, nstate / 2);

    __bf16* hin = hbf0;
    __bf16* hout = hbf1;
    for (int s = 0; s < SS; ++s) {
      lstm_step_k<<<dim3(BB / 16, HH / 16), 32, 0, stream>>>(
          hin, hout, WhhB, Gx, cbuf, hbuf, sc, s, store_sc);
      __bf16* tmp = hin; hin = hout; hout = tmp;
    }
  };

  run_rnn(x1, key_c, /*store_sc=*/1);   // sc = all hidden states of RNN1
  run_rnn(x2, key_r, /*store_sc=*/0);   // hbuf = final hidden state r of RNN2
  const float* r = hbuf;

  // -------- attention tail --------
  attn_u_k  <<<BB, HH, 0, stream>>>(r, W_attn, b_attn, u, e0);
  energies_k<<<dim3(BB, SS / 8), 256, 0, stream>>>(sc, u, e0, x1mask, energ);
  softmax_k <<<BB, 256, 0, stream>>>(energ, alpha);
  cattn_k   <<<BB, HH, 0, stream>>>(alpha, sc, cattn);
  final_k   <<<BB, HH, 0, stream>>>(cattn, Mw, r, b0, out);
}
